// FlexMultiHeadAttention_25881472926383
// MI455X (gfx1250) — matrix operand order repaired and checked by reading, not compiled, not run on hardware
//
#include <hip/hip_runtime.h>
#include <hip/hip_bf16.h>
#include <math.h>

// ---------------------------------------------------------------------------
// MI455X (gfx1250) fused multi-head attention forward.
//   S=2048, B=2, E=1024, H=16, hd=64. ~69 GFLOP, ~100MB HBM traffic.
//   bf16 WMMA (v_wmma_f32_16x16x32_bf16), f32 accumulation, flash-attention
//   middle stage. K/V staged per block into double-buffered LDS via gfx1250
//   GLOBAL_LOAD_ASYNC_TO_LDS_B128 (ASYNCcnt). GEMM + flash loops are 2x
//   unrolled so the register/LDS double buffers rotate structurally -- no
//   v_dual_mov register-copy chains in the steady state (round-3 asm showed
//   ~40 copies/iter from the single-buffer-rotation formulation).
// ---------------------------------------------------------------------------

typedef __bf16 bf16_t;
typedef __attribute__((ext_vector_type(16))) __bf16 v16bf;
typedef __attribute__((ext_vector_type(8)))  float  v8f;

static constexpr int kS  = 2048;
static constexpr int kB  = 2;
static constexpr int kE  = 1024;
static constexpr int kH  = 16;
static constexpr int kHD = 64;
static constexpr int kBH = kB * kH;
static constexpr float kLog2e = 1.44269504088896340736f;

__device__ __forceinline__ v8f wmma_bf16(v16bf a, v16bf b, v8f c) {
  return __builtin_amdgcn_wmma_f32_16x16x32_bf16(false, a, false, b,
                                                 (short)0, c, false, false);
}

__device__ __forceinline__ v8f zero8() {
  v8f z;
#pragma unroll
  for (int i = 0; i < 8; ++i) z[i] = 0.0f;
  return z;
}

// A fragment: 16x32 (MxK) bf16, row-major source, leading dim `ld`.
// lanes 0-15: M=lane, K={0..7,16..23}; lanes 16-31: M=lane-16, K={8..15,24..31}
__device__ __forceinline__ v16bf load_a_frag(const bf16_t* p0, int ld) {
  const int lane = threadIdx.x & 31;
  const int row  = lane & 15;
  const int koff = (lane & 16) ? 8 : 0;
  const bf16_t* p = p0 + (size_t)row * ld + koff;
  union { v16bf v; uint4 q[2]; } u;
  u.q[0] = *(const uint4*)(p);
  u.q[1] = *(const uint4*)(p + 16);
  return u.v;
}

// B fragment: 32x16 (KxN) bf16 from NT operand Bt[n][k], leading dim `ld`.
// lanes 0-15: N=lane, K={0..7,16..23}; lanes 16-31: N=lane-16, K={8..15,24..31}
__device__ __forceinline__ v16bf load_b_frag(const bf16_t* p0, int ld) {
  const int lane = threadIdx.x & 31;
  const int n    = lane & 15;
  const int koff = (lane & 16) ? 8 : 0;
  const bf16_t* p = p0 + (size_t)n * ld + koff;
  union { v16bf v; uint4 q[2]; } u;
  u.q[0] = *(const uint4*)(p);
  u.q[1] = *(const uint4*)(p + 16);
  return u.v;
}

__device__ __forceinline__ float half16_max(float x) {
  x = fmaxf(x, __shfl_xor(x, 1, 16));
  x = fmaxf(x, __shfl_xor(x, 2, 16));
  x = fmaxf(x, __shfl_xor(x, 4, 16));
  x = fmaxf(x, __shfl_xor(x, 8, 16));
  return x;
}
__device__ __forceinline__ float half16_sum(float x) {
  x += __shfl_xor(x, 1, 16);
  x += __shfl_xor(x, 2, 16);
  x += __shfl_xor(x, 4, 16);
  x += __shfl_xor(x, 8, 16);
  return x;
}

// ---------------- gfx1250 ASYNCcnt global->LDS path ------------------------
#if __has_builtin(__builtin_amdgcn_global_load_async_to_lds_b128)
#define HAS_ASYNC_LDS 1
typedef int gcc_v4i __attribute__((vector_size(16)));
typedef __attribute__((address_space(1))) gcc_v4i* as1_v4i_ptr;
typedef __attribute__((address_space(3))) gcc_v4i* as3_v4i_ptr;

__device__ __forceinline__ void async_cp16(bf16_t* lds_dst, const bf16_t* gsrc) {
  __builtin_amdgcn_global_load_async_to_lds_b128(
      (as1_v4i_ptr)gsrc, (as3_v4i_ptr)lds_dst, 0, 0);
}
__device__ __forceinline__ void wait_async0() {
#if __has_builtin(__builtin_amdgcn_s_wait_asynccnt)
  __builtin_amdgcn_s_wait_asynccnt(0);
#else
  asm volatile("s_wait_asynccnt 0" ::: "memory");
#endif
}
#else
#define HAS_ASYNC_LDS 0
#endif

// ------------------------------ kernels ------------------------------------

__global__ void __launch_bounds__(256) cvt_f32_bf16(const float* __restrict__ in,
                                                    bf16_t* __restrict__ out, int n) {
  for (int i = blockIdx.x * blockDim.x + threadIdx.x; i < n;
       i += gridDim.x * blockDim.x)
    out[i] = (bf16_t)in[i];
}

// ---- NT-GEMM core: 16x64 tile per wave, K pipelined 2x with structural
// ping/pong (compute (a0,b0) while (a1,b1) loads land, and vice versa). ----
__device__ __forceinline__ void gemm_16x64(const bf16_t* __restrict__ Ap,
                                           const bf16_t* __restrict__ Wp,
                                           v8f acc[4]) {
  v16bf a0 = load_a_frag(Ap, kE);
  v16bf b0[4];
#pragma unroll
  for (int j = 0; j < 4; ++j) b0[j] = load_b_frag(Wp + (size_t)j * 16 * kE, kE);

  for (int k = 0; k < kE; k += 64) {
    // phase 1: load k+32 while computing k
    v16bf a1 = load_a_frag(Ap + k + 32, kE);
    v16bf b1[4];
#pragma unroll
    for (int j = 0; j < 4; ++j)
      b1[j] = load_b_frag(Wp + (size_t)j * 16 * kE + k + 32, kE);
#pragma unroll
    for (int j = 0; j < 4; ++j) acc[j] = wmma_bf16(a0, b0[j], acc[j]);

    // phase 2: load k+64 (clamped to 0 on last iter, redundant but branchless)
    const int kn = (k + 64 < kE) ? k + 64 : 0;
    a0 = load_a_frag(Ap + kn, kE);
#pragma unroll
    for (int j = 0; j < 4; ++j)
      b0[j] = load_b_frag(Wp + (size_t)j * 16 * kE + kn, kE);
#pragma unroll
    for (int j = 0; j < 4; ++j) acc[j] = wmma_bf16(a1, b1[j], acc[j]);
  }
}

// qkv = Xbf @ Wqkv^T + bias; epilogue scatters Q(scaled)/K row-major (s,d)
// and V transposed (d,s), [BH]-major.
__global__ void __launch_bounds__(128) qkv_gemm(const bf16_t* __restrict__ X,
                                                const bf16_t* __restrict__ W,
                                                const float*  __restrict__ bias,
                                                bf16_t* __restrict__ Qb,
                                                bf16_t* __restrict__ Kb,
                                                bf16_t* __restrict__ Vt) {
  const int wave = threadIdx.x >> 5;
  const int lane = threadIdx.x & 31;
  const int rt = blockIdx.x;                 // 16-row tile of 4096 rows
  const int ct = blockIdx.y * 4 + wave;      // 64-col tile of 3072 cols

  v8f acc[4];
#pragma unroll
  for (int j = 0; j < 4; ++j) acc[j] = zero8();
  gemm_16x64(X + (size_t)rt * 16 * kE, W + (size_t)ct * 64 * kE, acc);

  const int row0 = (lane & 16) ? 8 : 0;
  const int col  = lane & 15;
#pragma unroll
  for (int j = 0; j < 4; ++j) {
    const int f = ct * 64 + j * 16 + col;
    const float bv = bias[f];
    const int region = f >> 10;        // 0:q 1:k 2:v (64-col tile never straddles)
    const int el = f & (kE - 1);
    const int h = el >> 6, d = el & 63;
#pragma unroll
    for (int r = 0; r < 8; ++r) {
      const int n = rt * 16 + row0 + r;
      const int s = n >> 1, bb = n & 1;
      const size_t bh = (size_t)(bb * kH + h);
      const float v = acc[j][r] + bv;
      if (region == 0)
        Qb[(bh * kS + s) * kHD + d] = (bf16_t)(v * 0.125f);  // hd^-0.5
      else if (region == 1)
        Kb[(bh * kS + s) * kHD + d] = (bf16_t)v;
      else
        Vt[(bh * kHD + d) * kS + s] = (bf16_t)v;
    }
  }
}

// ---- flash attention ----
// K tile: 32 keys x 64 d = contiguous 2048-elem chunk. 256 uint4, 2/thread.
// V tile: 64 d-rows x 32 s (ld=kS global, ld=32 LDS). 256 uint4, 2/thread.
__device__ __forceinline__ void stage_issue(const bf16_t* Kh, const bf16_t* Vth,
                                            int kt, uint4* kr, uint4* vr) {
  const int t = threadIdx.x;
  const uint4* ksrc = (const uint4*)(Kh + (size_t)kt * kHD);
  kr[0] = ksrc[t];
  kr[1] = ksrc[t + 128];
  const int c0 = t, c1 = t + 128;
  vr[0] = *(const uint4*)(Vth + (size_t)(c0 >> 2) * kS + kt + (c0 & 3) * 8);
  vr[1] = *(const uint4*)(Vth + (size_t)(c1 >> 2) * kS + kt + (c1 & 3) * 8);
}
__device__ __forceinline__ void stage_commit(bf16_t* Ks, bf16_t* Vs,
                                             const uint4* kr, const uint4* vr) {
  const int t = threadIdx.x;
  ((uint4*)Ks)[t] = kr[0];
  ((uint4*)Ks)[t + 128] = kr[1];
  const int c0 = t, c1 = t + 128;
  *(uint4*)(Vs + (c0 >> 2) * 32 + (c0 & 3) * 8) = vr[0];
  *(uint4*)(Vs + (c1 >> 2) * 32 + (c1 & 3) * 8) = vr[1];
}
#if HAS_ASYNC_LDS
__device__ __forceinline__ void stage_async(const bf16_t* Kh, const bf16_t* Vth,
                                            int kt, bf16_t* Ks, bf16_t* Vs) {
  const int t = threadIdx.x;
  const bf16_t* kp = Kh + (size_t)kt * kHD;
  async_cp16(Ks + t * 8, kp + t * 8);
  async_cp16(Ks + (t + 128) * 8, kp + (t + 128) * 8);
  const int c0 = t, c1 = t + 128;
  async_cp16(Vs + (c0 >> 2) * 32 + (c0 & 3) * 8,
             Vth + (size_t)(c0 >> 2) * kS + kt + (c0 & 3) * 8);
  async_cp16(Vs + (c1 >> 2) * 32 + (c1 & 3) * 8,
             Vth + (size_t)(c1 >> 2) * kS + kt + (c1 & 3) * 8);
}
#endif

struct FaState {
  v16bf qa0, qa1;
  v8f acc[4];
  float m[8], l[8];
};

// One 32-key step: stage next tile into (Ksn,Vsn) [ASYNCcnt DMA], compute
// scores + online softmax + P.V from (Kc,Vc), then barrier hand-off.
__device__ __forceinline__ void fa_step(FaState& st,
                                        const bf16_t* Kc, const bf16_t* Vc,
                                        bf16_t* Ksn, bf16_t* Vsn, int ktn,
                                        const bf16_t* Kh, const bf16_t* Vth,
                                        bf16_t* P, int row0, int col) {
#if HAS_ASYNC_LDS
  stage_async(Kh, Vth, ktn, Ksn, Vsn);
#else
  uint4 kr[2], vr[2];
  stage_issue(Kh, Vth, ktn, kr, vr);
#endif

  v16bf kb00 = load_b_frag(Kc + 0,            64);
  v16bf kb01 = load_b_frag(Kc + 32,           64);
  v16bf kb10 = load_b_frag(Kc + 16 * 64 + 0,  64);
  v16bf kb11 = load_b_frag(Kc + 16 * 64 + 32, 64);

  v8f s0 = zero8(), s1 = zero8();
  s0 = wmma_bf16(st.qa0, kb00, s0);
  s0 = wmma_bf16(st.qa1, kb01, s0);
  s1 = wmma_bf16(st.qa0, kb10, s1);
  s1 = wmma_bf16(st.qa1, kb11, s1);

  float p0[8], p1[8];
#pragma unroll
  for (int r = 0; r < 8; ++r) {
    float mx = half16_max(fmaxf(s0[r], s1[r]));
    float mn = fmaxf(st.m[r], mx);
    float corr = exp2f((st.m[r] - mn) * kLog2e);
    p0[r] = exp2f((s0[r] - mn) * kLog2e);
    p1[r] = exp2f((s1[r] - mn) * kLog2e);
    float rs = half16_sum(p0[r] + p1[r]);
    st.l[r] = st.l[r] * corr + rs;
    st.m[r] = mn;
    st.acc[0][r] *= corr; st.acc[1][r] *= corr;
    st.acc[2][r] *= corr; st.acc[3][r] *= corr;
  }

  // C-layout -> LDS (row-major 16x32) -> A-layout reload
#pragma unroll
  for (int r = 0; r < 8; ++r) {
    P[(row0 + r) * 32 + col]      = (bf16_t)p0[r];
    P[(row0 + r) * 32 + col + 16] = (bf16_t)p1[r];
  }
  asm volatile("s_wait_dscnt 0" ::: "memory");
  v16bf pa = load_a_frag(P, 32);

#pragma unroll
  for (int j = 0; j < 4; ++j) {
    v16bf vb = load_b_frag(Vc + j * 16 * 32, 32);
    st.acc[j] = wmma_bf16(pa, vb, st.acc[j]);
  }

#if HAS_ASYNC_LDS
  wait_async0();
#else
  stage_commit(Ksn, Vsn, kr, vr);
#endif
  __syncthreads();
}

__global__ void __launch_bounds__(128) flash_attn(const bf16_t* __restrict__ Qb,
                                                  const bf16_t* __restrict__ Kb,
                                                  const bf16_t* __restrict__ Vt,
                                                  bf16_t* __restrict__ O) {
  const int wave = threadIdx.x >> 5;
  const int lane = threadIdx.x & 31;
  const int bh = blockIdx.y;
  const int qt = blockIdx.x * 4 + wave;      // 16-row query tile, 0..127
  const int b = bh >> 4, h = bh & 15;

  const bf16_t* Qh  = Qb + (size_t)bh * kS * kHD;
  const bf16_t* Kh  = Kb + (size_t)bh * kS * kHD;
  const bf16_t* Vth = Vt + (size_t)bh * kHD * kS;

  __shared__ bf16_t Ks[2][32 * 64];   // ping/pong key tiles (2 x 4KB)
  __shared__ bf16_t Vs[2][64 * 32];   // ping/pong V^T tiles (2 x 4KB, ld=32)
  __shared__ bf16_t Pls[4][16 * 32];  // per-wave P staging (C->A relayout)
  bf16_t* P = Pls[wave];

  FaState st;
  st.qa0 = load_a_frag(Qh + (size_t)qt * 16 * kHD + 0,  kHD);
  st.qa1 = load_a_frag(Qh + (size_t)qt * 16 * kHD + 32, kHD);
#pragma unroll
  for (int r = 0; r < 8; ++r) { st.m[r] = -1e30f; st.l[r] = 0.0f; }
#pragma unroll
  for (int j = 0; j < 4; ++j) st.acc[j] = zero8();

  const int row0 = (lane & 16) ? 8 : 0;
  const int col  = lane & 15;
  const int nIter = kS / 32;                 // 64 (even)

  // prologue: stage tile 0 into buffer 0
#if HAS_ASYNC_LDS
  stage_async(Kh, Vth, 0, Ks[0], Vs[0]);
  wait_async0();
#else
  {
    uint4 kr[2], vr[2];
    stage_issue(Kh, Vth, 0, kr, vr);
    stage_commit(Ks[0], Vs[0], kr, vr);
  }
#endif
  __syncthreads();

  // structural ping/pong, 2 steps per trip (no cur/nxt selects, no copies)
  for (int it = 0; it < nIter; it += 2) {
    const int kt1 = (it + 1) * 32;                              // always < kS
    const int kt2 = (it + 2 < nIter) ? (it + 2) * 32 : 0;       // clamped
    fa_step(st, Ks[0], Vs[0], Ks[1], Vs[1], kt1, Kh, Vth, P, row0, col);
    fa_step(st, Ks[1], Vs[1], Ks[0], Vs[0], kt2, Kh, Vth, P, row0, col);
  }

  float inv[8];
#pragma unroll
  for (int r = 0; r < 8; ++r) inv[r] = 1.0f / st.l[r];

#pragma unroll
  for (int j = 0; j < 4; ++j) {
#pragma unroll
    for (int r = 0; r < 8; ++r) {
      const int s = qt * 16 + row0 + r;
      const int e = h * kHD + j * 16 + col;
      O[((size_t)s * kB + b) * kE + e] = (bf16_t)(st.acc[j][r] * inv[r]);
    }
  }
}

// out = AO @ Wout^T + bias (fp32 result).
__global__ void __launch_bounds__(128) out_proj(const bf16_t* __restrict__ A,
                                                const bf16_t* __restrict__ W,
                                                const float*  __restrict__ bias,
                                                float* __restrict__ out) {
  const int wave = threadIdx.x >> 5;
  const int lane = threadIdx.x & 31;
  const int rt = blockIdx.x;
  const int ct = blockIdx.y * 4 + wave;

  v8f acc[4];
#pragma unroll
  for (int j = 0; j < 4; ++j) acc[j] = zero8();
  gemm_16x64(A + (size_t)rt * 16 * kE, W + (size_t)ct * 64 * kE, acc);

  const int row0 = (lane & 16) ? 8 : 0;
  const int col  = lane & 15;
#pragma unroll
  for (int j = 0; j < 4; ++j) {
    const int f = ct * 64 + j * 16 + col;
    const float bv = bias[f];
#pragma unroll
    for (int r = 0; r < 8; ++r) {
      const int n = rt * 16 + row0 + r;
      out[(size_t)n * kE + f] = acc[j][r] + bv;
    }
  }
}

// ------------------------------ launcher -----------------------------------

extern "C" void kernel_launch(void* const* d_in, const int* in_sizes, int n_in,
                              void* d_out, int out_size, void* d_ws, size_t ws_size,
                              hipStream_t stream) {
  const float* q     = (const float*)d_in[0];   // (S,B,E)
  const float* w_in  = (const float*)d_in[1];   // (3E,E)
  const float* b_in  = (const float*)d_in[2];   // (3E,)
  const float* w_out = (const float*)d_in[3];   // (E,E)
  const float* b_out = (const float*)d_in[4];   // (E,)
  float* out = (float*)d_out;

  bf16_t* ws = (bf16_t*)d_ws;
  size_t o = 0;
  bf16_t* Xbf  = ws + o; o += (size_t)kS * kB * kE;     // 4096x1024
  bf16_t* Wqkv = ws + o; o += (size_t)3 * kE * kE;      // 3072x1024
  bf16_t* Wout = ws + o; o += (size_t)kE * kE;          // 1024x1024
  bf16_t* Qb   = ws + o; o += (size_t)kBH * kS * kHD;   // (BH,S,hd)
  bf16_t* Kb   = ws + o; o += (size_t)kBH * kS * kHD;   // (BH,S,hd)
  bf16_t* Vt   = ws + o; o += (size_t)kBH * kHD * kS;   // (BH,hd,S)
  bf16_t* AO   = ws + o;                                 // 4096x1024

  cvt_f32_bf16<<<2048, 256, 0, stream>>>(q,     Xbf,  kS * kB * kE);
  cvt_f32_bf16<<<2048, 256, 0, stream>>>(w_in,  Wqkv, 3 * kE * kE);
  cvt_f32_bf16<<<1024, 256, 0, stream>>>(w_out, Wout, kE * kE);

  qkv_gemm<<<dim3(256, 12), 128, 0, stream>>>(Xbf, Wqkv, b_in, Qb, Kb, Vt);
  flash_attn<<<dim3(32, 32), 128, 0, stream>>>(Qb, Kb, Vt, AO);
  out_proj<<<dim3(256, 4), 128, 0, stream>>>(AO, Wout, b_out, out);
}
